// CausalSelfAttention_75256416961143
// MI455X (gfx1250) — matrix operand order repaired and checked by reading, not compiled, not run on hardware
//
#include <hip/hip_runtime.h>
#include <hip/hip_bf16.h>
#include <stdint.h>

#define Tt 2048
#define Dd 1024
#define Hh 16
#define HD 64
#define BT 8192   // B*T

typedef __attribute__((ext_vector_type(16))) __bf16 v16bf;
typedef __attribute__((ext_vector_type(8)))  __bf16 v8bf;
typedef __attribute__((ext_vector_type(8)))  float  v8f;
typedef __attribute__((ext_vector_type(4)))  int    v4i;

#if defined(__AMDGCN__) && __has_builtin(__builtin_amdgcn_global_load_async_to_lds_b128)
#define HAS_ASYNC 1
#else
#define HAS_ASYNC 0
#endif

// Per-lane async 16B copy global -> LDS (ASYNCcnt-tracked on gfx1250).
__device__ __forceinline__ void async_ld16(void* lds, const void* g) {
#if HAS_ASYNC
    __builtin_amdgcn_global_load_async_to_lds_b128(
        (__attribute__((address_space(1))) v4i*)(uintptr_t)g,
        (__attribute__((address_space(3))) v4i*)(unsigned)(uintptr_t)lds,
        0, 0);
#else
    *(v8bf*)lds = *(const v8bf*)g;
#endif
}

template<int N>
__device__ __forceinline__ void wait_asynccnt() {
#if HAS_ASYNC
#if __has_builtin(__builtin_amdgcn_s_wait_asynccnt)
    __builtin_amdgcn_s_wait_asynccnt(N);
#else
    asm volatile("s_wait_asynccnt %0" :: "i"(N) : "memory");
#endif
#endif
}

__device__ __forceinline__ v16bf cat16(v8bf lo, v8bf hi) {
    return __builtin_shufflevector(lo, hi, 0,1,2,3,4,5,6,7,8,9,10,11,12,13,14,15);
}

__device__ __forceinline__ v8f wmma_bf16(v16bf a, v16bf b, v8f c) {
    return __builtin_amdgcn_wmma_f32_16x16x32_bf16(false, a, false, b, (short)0, c, false, false);
}

// A-operand fragment (16x32, MxK) from LDS [rows][PITCH] row-major.
template<int PITCH>
__device__ __forceinline__ v16bf lds_a_frag(const __bf16* base, int row0, int kbase, int lane) {
    int m  = lane & 15;
    int c0 = kbase + ((lane < 16) ? 0 : 8);
    const __bf16* p = base + (row0 + m) * PITCH + c0;
    return cat16(*(const v8bf*)p, *(const v8bf*)(p + 16));
}

// B-operand fragment (32x16, KxN) from LDS stored [n][k].
template<int PITCH>
__device__ __forceinline__ v16bf lds_b_frag(const __bf16* base, int n0, int kbase, int lane) {
    int n  = lane & 15;
    int k0 = kbase + ((lane < 16) ? 0 : 8);
    const __bf16* p = base + (n0 + n) * PITCH + k0;
    return cat16(*(const v8bf*)p, *(const v8bf*)(p + 16));
}

__global__ void cvt_f32_bf16(const float* __restrict__ in, __bf16* __restrict__ out, int n) {
    int i = blockIdx.x * blockDim.x + threadIdx.x;
    int stride = gridDim.x * blockDim.x;
    for (; i < n; i += stride) out[i] = (__bf16)in[i];
}

// f32 [K][N] row-major -> bf16 [N][K] row-major (weights pre-transpose, done once).
__global__ void cvt_transpose(const float* __restrict__ in, __bf16* __restrict__ out,
                              int Ndim, int Kdim) {
    int n = blockIdx.x * blockDim.x + threadIdx.x;
    int k = blockIdx.y;
    if (n < Ndim) out[(size_t)n * Kdim + k] = (__bf16)in[(size_t)k * Ndim + n];
}

// 128x128 tile GEMM, BK=64, double-buffered async LDS staging.
template<int N, int K, int MODE>
__global__ __launch_bounds__(256) void gemm_bf16_kernel(
    const __bf16* __restrict__ A, const __bf16* __restrict__ Bt,
    const float* __restrict__ bias,
    __bf16* __restrict__ Qo, __bf16* __restrict__ Ko, __bf16* __restrict__ Vo,
    float* __restrict__ Fo)
{
    __shared__ __bf16 As[2][128][72];
    __shared__ __bf16 Bs[2][128][72];

    int tid = threadIdx.x, lane = tid & 31, wave = tid >> 5;
    int wm = wave >> 2, wn = wave & 3;
    int m0 = blockIdx.x * 128, n0 = blockIdx.y * 128;

    auto stage = [&](int buf, int kt) {
        int row   = tid >> 1;
        int cbase = (tid & 1) * 4;
        #pragma unroll
        for (int i = 0; i < 4; ++i) {
            int ch = cbase + i;
            async_ld16(&As[buf][row][ch * 8], &A [(size_t)(m0 + row) * K + kt + ch * 8]);
            async_ld16(&Bs[buf][row][ch * 8], &Bt[(size_t)(n0 + row) * K + kt + ch * 8]);
        }
    };

    v8f acc[4][2];
    #pragma unroll
    for (int mi = 0; mi < 4; ++mi)
        #pragma unroll
        for (int ni = 0; ni < 2; ++ni) acc[mi][ni] = {};

    stage(0, 0);
    #pragma unroll 1
    for (int kt = 0; kt < K; kt += 64) {
        int cur = (kt >> 6) & 1;
        if (kt + 64 < K) {
            stage(cur ^ 1, kt + 64);
            wait_asynccnt<8>();
        } else {
            wait_asynccnt<0>();
        }
        __syncthreads();

        #pragma unroll
        for (int ks = 0; ks < 64; ks += 32) {
            v16bf af[4], bf[2];
            #pragma unroll
            for (int mi = 0; mi < 4; ++mi)
                af[mi] = lds_a_frag<72>(&As[cur][0][0], wm * 64 + mi * 16, ks, lane);
            #pragma unroll
            for (int ni = 0; ni < 2; ++ni)
                bf[ni] = lds_b_frag<72>(&Bs[cur][0][0], wn * 32 + ni * 16, ks, lane);
            #pragma unroll
            for (int mi = 0; mi < 4; ++mi)
                #pragma unroll
                for (int ni = 0; ni < 2; ++ni)
                    acc[mi][ni] = wmma_bf16(af[mi], bf[ni], acc[mi][ni]);
        }
        __syncthreads();
    }

    int rowadd = (lane < 16) ? 0 : 8;
    #pragma unroll
    for (int ni = 0; ni < 2; ++ni) {
        int col = n0 + wn * 32 + ni * 16 + (lane & 15);
        float bv = bias[col];
        #pragma unroll
        for (int mi = 0; mi < 4; ++mi) {
            #pragma unroll
            for (int r = 0; r < 8; ++r) {
                int grow = m0 + wm * 64 + mi * 16 + r + rowadd;
                float val = acc[mi][ni][r] + bv;
                if constexpr (MODE == 0) {
                    int which = col >> 10, d = col & 1023;
                    int h = d >> 6, hd = d & 63;
                    int b = grow >> 11, t = grow & 2047;
                    if (which == 0)
                        Qo[(((size_t)(b * Hh + h)) * Tt + t) * HD + hd] = (__bf16)(val * 0.125f);
                    else if (which == 1)
                        Ko[(((size_t)(b * Hh + h)) * Tt + t) * HD + hd] = (__bf16)val;
                    else  // V stored head-transposed [B,H,HD,T]
                        Vo[(((size_t)(b * Hh + h)) * HD + hd) * Tt + t] = (__bf16)val;
                } else {
                    Fo[(size_t)grow * N + col] = val;
                }
            }
        }
    }
}

// Flash attention: 128 threads (4 waves), each wave owns 32 q-rows (2x16 sub-
// tiles sharing K/V fragments -> 16 WMMA per 32-token iteration), block covers
// 128 q-rows. Double-buffered async K/V staging, online softmax in C layout.
__global__ __launch_bounds__(128) void attn_kernel(
    const __bf16* __restrict__ Q, const __bf16* __restrict__ Kt,
    const __bf16* __restrict__ Vt, __bf16* __restrict__ Y)
{
    __shared__ __bf16 Ks[2][32][72];        // [token][hd]
    __shared__ __bf16 Vs[2][64][40];        // [hd][token]
    __shared__ __bf16 Ps[4][2][16][40];     // per-wave, per-qtile P staging

    int tid = threadIdx.x, lane = tid & 31, wave = tid >> 5;
    int q0 = blockIdx.x * 128;
    int h = blockIdx.y, b = blockIdx.z;
    const __bf16* qb = Q  + ((size_t)(b * Hh + h)) * Tt * HD;
    const __bf16* kb = Kt + ((size_t)(b * Hh + h)) * Tt * HD;
    const __bf16* vt = Vt + ((size_t)(b * Hh + h)) * HD * Tt;
    int qw = q0 + wave * 32;

    auto stage = [&](int buf, int kt) {
        #pragma unroll
        for (int i = 0; i < 2; ++i) {       // K tile: 32 tok x 64 hd
            int tok = (tid >> 3) + i * 16;
            int ch  = tid & 7;
            async_ld16(&Ks[buf][tok][ch * 8], &kb[(size_t)(kt + tok) * HD + ch * 8]);
        }
        {
            int hd = tid >> 1;              // V tile: 64 hd x 32 tok
            int cb = (tid & 1) * 2;
            #pragma unroll
            for (int i = 0; i < 2; ++i) {
                int ch = cb + i;
                async_ld16(&Vs[buf][hd][ch * 8], &vt[(size_t)hd * Tt + kt + ch * 8]);
            }
        }
    };

    // Q fragments: 2 q-subtiles x 2 K-steps
    int m  = lane & 15;
    int c0 = (lane < 16) ? 0 : 8;
    v16bf qf[2][2];
    #pragma unroll
    for (int qt = 0; qt < 2; ++qt)
        #pragma unroll
        for (int s = 0; s < 2; ++s) {
            const __bf16* p = qb + (size_t)(qw + qt * 16 + m) * HD + s * 32 + c0;
            qf[qt][s] = cat16(*(const v8bf*)p, *(const v8bf*)(p + 16));
        }

    v8f o[2][4];
    float mrow[2][8], lrow[2][8];
    const float NEG_INF = -__builtin_inff();
    #pragma unroll
    for (int qt = 0; qt < 2; ++qt) {
        #pragma unroll
        for (int c = 0; c < 4; ++c) o[qt][c] = {};
        #pragma unroll
        for (int r = 0; r < 8; ++r) { mrow[qt][r] = NEG_INF; lrow[qt][r] = 0.0f; }
    }

    int rowadd = (lane < 16) ? 0 : 8;
    int ktend = q0 + 128;

    stage(0, 0);
    for (int kt = 0; kt < ktend; kt += 32) {
        int cur = (kt >> 5) & 1;
        if (kt + 32 < ktend) {
            stage(cur ^ 1, kt + 32);
            wait_asynccnt<4>();
        } else {
            wait_asynccnt<0>();
        }
        __syncthreads();

        // S = Q K^T : K-fragments loaded once, reused by both q-subtiles
        v8f s0[2], s1[2];
        s0[0] = {}; s0[1] = {}; s1[0] = {}; s1[1] = {};
        #pragma unroll
        for (int s = 0; s < 2; ++s) {
            v16bf bk0 = lds_b_frag<72>(&Ks[cur][0][0], 0,  s * 32, lane);
            v16bf bk1 = lds_b_frag<72>(&Ks[cur][0][0], 16, s * 32, lane);
            #pragma unroll
            for (int qt = 0; qt < 2; ++qt) {
                s0[qt] = wmma_bf16(qf[qt][s], bk0, s0[qt]);
                s1[qt] = wmma_bf16(qf[qt][s], bk1, s1[qt]);
            }
        }

        // Online softmax per q-subtile
        int colb = kt + (lane & 15);
        #pragma unroll
        for (int qt = 0; qt < 2; ++qt) {
            #pragma unroll
            for (int r = 0; r < 8; ++r) {
                int row = qw + qt * 16 + rowadd + r;
                float a0 = (colb      <= row) ? s0[qt][r] : NEG_INF;
                float a1 = (colb + 16 <= row) ? s1[qt][r] : NEG_INF;
                float mx = fmaxf(a0, a1);
                #pragma unroll
                for (int off = 8; off > 0; off >>= 1) mx = fmaxf(mx, __shfl_xor(mx, off, 16));
                float mnew = fmaxf(mrow[qt][r], mx);
                float al = __expf(mrow[qt][r] - mnew);
                float p0 = __expf(a0 - mnew);
                float p1 = __expf(a1 - mnew);
                float sum = p0 + p1;
                #pragma unroll
                for (int off = 8; off > 0; off >>= 1) sum += __shfl_xor(sum, off, 16);
                lrow[qt][r] = lrow[qt][r] * al + sum;
                mrow[qt][r] = mnew;
                #pragma unroll
                for (int c = 0; c < 4; ++c) o[qt][c][r] *= al;
                Ps[wave][qt][rowadd + r][lane & 15]        = (__bf16)p0;
                Ps[wave][qt][rowadd + r][16 + (lane & 15)] = (__bf16)p1;
            }
        }

        // Intra-wave cross-lane LDS RAW on P staging
        asm volatile("s_wait_dscnt 0" ::: "memory");

        v16bf pf[2];
        #pragma unroll
        for (int qt = 0; qt < 2; ++qt)
            pf[qt] = lds_a_frag<40>(&Ps[wave][qt][0][0], 0, 0, lane);
        #pragma unroll
        for (int c = 0; c < 4; ++c) {
            v16bf vf = lds_b_frag<40>(&Vs[cur][0][0], c * 16, 0, lane);
            #pragma unroll
            for (int qt = 0; qt < 2; ++qt)
                o[qt][c] = wmma_bf16(pf[qt], vf, o[qt][c]);
        }
        __syncthreads();
    }

    size_t yb = ((size_t)b * Tt) * Dd + (size_t)h * HD;
    #pragma unroll
    for (int qt = 0; qt < 2; ++qt) {
        #pragma unroll
        for (int r = 0; r < 8; ++r) {
            float inv = 1.0f / lrow[qt][r];
            #pragma unroll
            for (int c = 0; c < 4; ++c) o[qt][c][r] *= inv;
        }
        #pragma unroll
        for (int c = 0; c < 4; ++c) {
            #pragma unroll
            for (int r = 0; r < 8; ++r) {
                int t = qw + qt * 16 + rowadd + r;
                Y[yb + (size_t)t * Dd + c * 16 + (lane & 15)] = (__bf16)o[qt][c][r];
            }
        }
    }
}

extern "C" void kernel_launch(void* const* d_in, const int* in_sizes, int n_in,
                              void* d_out, int out_size, void* d_ws, size_t ws_size,
                              hipStream_t stream) {
    const float* x      = (const float*)d_in[0];
    const float* W_qkv  = (const float*)d_in[1];
    const float* b_qkv  = (const float*)d_in[2];
    const float* W_proj = (const float*)d_in[3];
    const float* b_proj = (const float*)d_in[4];
    float* out = (float*)d_out;

    char* ws = (char*)d_ws;
    const size_t XB = (size_t)BT * Dd;
    __bf16* xb     = (__bf16*)ws;  ws += XB * 2;
    __bf16* wqkvT  = (__bf16*)ws;  ws += (size_t)3 * Dd * Dd * 2;   // [3D, D]
    __bf16* wprojT = (__bf16*)ws;  ws += (size_t)Dd * Dd * 2;       // [D, D]
    __bf16* q      = (__bf16*)ws;  ws += XB * 2;                     // [B,H,T,HD]
    __bf16* k      = (__bf16*)ws;  ws += XB * 2;                     // [B,H,T,HD]
    __bf16* vT     = (__bf16*)ws;  ws += XB * 2;                     // [B,H,HD,T]
    __bf16* yatt   = (__bf16*)ws;                                    // [B*T, D]

    cvt_f32_bf16<<<2048, 256, 0, stream>>>(x, xb, (int)XB);
    cvt_transpose<<<dim3(3 * Dd / 256, Dd), 256, 0, stream>>>(W_qkv, wqkvT, 3 * Dd, Dd);
    cvt_transpose<<<dim3(Dd / 256, Dd), 256, 0, stream>>>(W_proj, wprojT, Dd, Dd);

    // QKV: [8192,1024] x [1024,3072]
    gemm_bf16_kernel<3 * Dd, Dd, 0><<<dim3(BT / 128, 3 * Dd / 128), 256, 0, stream>>>(
        xb, wqkvT, b_qkv, q, k, vT, nullptr);

    // Attention: grid (T/128, H, B), 128 q-rows per block
    attn_kernel<<<dim3(Tt / 128, Hh, 4), 128, 0, stream>>>(q, k, vT, yatt);

    // Projection: [8192,1024] x [1024,1024] -> f32 out
    gemm_bf16_kernel<Dd, Dd, 1><<<dim3(BT / 128, Dd / 128), 256, 0, stream>>>(
        yatt, wprojT, b_proj, nullptr, nullptr, nullptr, out);
}
